// LinearTransformerExpert_18425409699870
// MI455X (gfx1250) — matrix operand order repaired and checked by reading, not compiled, not run on hardware
//
#include <hip/hip_runtime.h>
#include <hip/hip_bf16.h>

// ---------------------------------------------------------------------------
// MI455X (gfx1250): all matrix math via v_wmma_f32_16x16x32_bf16 (f32 acc).
// Weights converted f32 -> bf16 and transposed once ([K,N] -> [N,K]).
// GEMM tiles are moved global -> LDS with GLOBAL_LOAD_ASYNC_TO_LDS_B128
// (ASYNCcnt-tracked, no VGPR staging, no spills), WMMA reads via ds_load_b128.
// ---------------------------------------------------------------------------

typedef __attribute__((ext_vector_type(16))) __bf16 bf16x16;
typedef __attribute__((ext_vector_type(8)))  float  f32x8;

__device__ __forceinline__ unsigned short f32_to_bf16(float f) {
    unsigned int u = __float_as_uint(f);
    unsigned int r = u + 0x7FFFu + ((u >> 16) & 1u);   // round-to-nearest-even
    return (unsigned short)(r >> 16);
}
__device__ __forceinline__ float bf16_to_f32(unsigned short u) {
    return __uint_as_float(((unsigned int)u) << 16);
}

// ---------------------------------------------------------------------------
// Elementwise: build xs = concat(x, rwkv_state) in bf16, and x in bf16.
// ---------------------------------------------------------------------------
__global__ __launch_bounds__(256)
void convert_concat_kernel(const float* __restrict__ x,
                           const float* __restrict__ st,
                           unsigned short* __restrict__ xs_bf16,
                           unsigned short* __restrict__ x_bf16,
                           int N, int C) {
    size_t i = (size_t)blockIdx.x * blockDim.x + threadIdx.x;   // over N*2C
    size_t total = (size_t)N * 2 * C;
    if (i >= total) return;
    int twoC = 2 * C;
    size_t n = i / twoC;
    int    j = (int)(i - n * twoC);
    if (j < C) {
        float v = x[n * C + j];
        unsigned short b = f32_to_bf16(v);
        xs_bf16[i] = b;
        x_bf16[n * C + j] = b;
    } else {
        xs_bf16[i] = f32_to_bf16(st[n * C + (j - C)]);
    }
}

// ---------------------------------------------------------------------------
// One-time weight prep: W f32 [K,N] row-major -> Wt bf16 [N,K] row-major.
// ---------------------------------------------------------------------------
__global__ __launch_bounds__(256)
void transpose_to_bf16_kernel(const float* __restrict__ W,
                              unsigned short* __restrict__ Wt,
                              int K, int N) {
    __shared__ float tile[32][33];
    const int k0 = blockIdx.y * 32;
    const int n0 = blockIdx.x * 32;
    const int tx = threadIdx.x & 31;
    const int ty = threadIdx.x >> 5;        // 0..7
#pragma unroll
    for (int i = 0; i < 4; ++i) {
        int r = ty + i * 8;
        tile[r][tx] = W[(size_t)(k0 + r) * N + n0 + tx];
    }
    __syncthreads();
#pragma unroll
    for (int i = 0; i < 4; ++i) {
        int r = ty + i * 8;                 // n within tile
        Wt[(size_t)(n0 + r) * K + k0 + tx] = f32_to_bf16(tile[tx][r]);
    }
}

// ---------------------------------------------------------------------------
// Tiled bf16 WMMA GEMM: C[M,N] = A[M,K] @ B[K,N], B supplied pre-transposed
// as Bt bf16 [N,K]. Block tile 128x256, BK=32, 256 threads = 8 waves (2x4);
// each wave computes 64x64 = 4x4 accumulators (16 WMMA per K-step).
// Tile movement: global_load_async_to_lds_b128 (ASYNCcnt), double-buffered.
// ---------------------------------------------------------------------------
constexpr int EPI_BF16     = 0;   // store bf16
constexpr int EPI_F32_BIAS = 1;   // store f32 + bias[col]
constexpr int EPI_RECON    = 2;   // (val - aux[idx])^2 -> atomicAdd(loss)
constexpr int EPI_GATE     = 3;   // val * sigmoid(aux[idx]) -> f32

#define LDK 40   // padded LDS row stride (ushorts); 80B rows, 16B aligned

template <int EPI>
__global__ __launch_bounds__(256)
void gemm_bf16_wmma(const unsigned short* __restrict__ A,  int lda,
                    const unsigned short* __restrict__ Bt, int ldbt, // [N,K]
                    void* __restrict__ Cout, int ldc,
                    const float* __restrict__ aux,
                    float* __restrict__ lossAccum,
                    int K) {
    __shared__ __align__(16) unsigned short ldsA[2][128 * LDK];
    __shared__ __align__(16) unsigned short ldsB[2][256 * LDK];

    const int tid  = threadIdx.x;
    const int lane = tid & 31;
    const int wave = tid >> 5;
    const int wm   = wave & 1;          // 0..1 -> 64-row stripe
    const int wn   = wave >> 1;         // 0..3 -> 64-col stripe
    const int kg   = lane >> 4;         // half-wave K-group
    const int l16  = lane & 15;

    const int blockCol = blockIdx.x * 256;
    const int blockRow = blockIdx.y * 128;

    f32x8 acc[4][4];
#pragma unroll
    for (int i = 0; i < 4; ++i)
#pragma unroll
        for (int j = 0; j < 4; ++j)
#pragma unroll
            for (int e = 0; e < 8; ++e) acc[i][j][e] = 0.0f;

    const int NK = K >> 5;   // K / 32

    // 32-bit LDS offsets of the staging buffers (block-relative).
    const unsigned ldsAbase = (unsigned)(size_t)&ldsA[0][0];
    const unsigned ldsBbase = (unsigned)(size_t)&ldsB[0][0];

    // Async DMA one 128x32 A tile + 256x32 B tile into LDS buffer `buf`.
    // Each lane moves 16B per issue: 2 issues for A, 4 for B.
    auto issueAsync = [&](int kt, int buf) {
#pragma unroll
        for (int i = 0; i < 2; ++i) {
            int c   = tid + i * 256;            // chunk id 0..511
            int row = c >> 2;                   // 0..127
            int sub = (c & 3) * 8;              // ushort offset within row
            const unsigned short* g =
                A + (size_t)(blockRow + row) * lda + (size_t)kt * 32 + sub;
            unsigned l = ldsAbase + (unsigned)(buf * (128 * LDK) + row * LDK + sub) * 2u;
            asm volatile("global_load_async_to_lds_b128 %0, %1, off"
                         :: "v"(l), "v"(g) : "memory");
            if (i == 0 && kt + 2 < NK) __builtin_prefetch(g + 64, 0, 1);
        }
#pragma unroll
        for (int i = 0; i < 4; ++i) {
            int c   = tid + i * 256;            // chunk id 0..1023
            int row = c >> 2;                   // 0..255
            int sub = (c & 3) * 8;
            const unsigned short* g =
                Bt + (size_t)(blockCol + row) * ldbt + (size_t)kt * 32 + sub;
            unsigned l = ldsBbase + (unsigned)(buf * (256 * LDK) + row * LDK + sub) * 2u;
            asm volatile("global_load_async_to_lds_b128 %0, %1, off"
                         :: "v"(l), "v"(g) : "memory");
            if (i == 0 && kt + 2 < NK) __builtin_prefetch(g + 64, 0, 1);
        }
    };

    auto ldfrag = [&](const unsigned short* base, int o0, int o1) -> bf16x16 {
        union { uint4 u[2]; bf16x16 v; } f;
        f.u[0] = *(const uint4*)(base + o0);
        f.u[1] = *(const uint4*)(base + o1);
        return f.v;
    };

    auto compute = [&](int buf) {
        const unsigned short* lA = ldsA[buf];
        const unsigned short* lB = ldsB[buf];
        bf16x16 bf[4];
#pragma unroll
        for (int j = 0; j < 4; ++j) {
            // ISA B 32x16 bf16 layout: lane<16 holds K {0..7,16..23} of column l16
            int col = wn * 64 + j * 16 + l16;
            int o0  = col * LDK + 8 * kg;
            bf[j] = ldfrag(lB, o0, o0 + 16);
        }
#pragma unroll
        for (int i = 0; i < 4; ++i) {
            // ISA A 16x32 bf16 layout: lane<16 holds K {0..7,16..23} of row l16
            int row = wm * 64 + i * 16 + l16;
            int o0  = row * LDK + 8 * kg;
            bf16x16 af = ldfrag(lA, o0, o0 + 16);
#pragma unroll
            for (int j = 0; j < 4; ++j)
                acc[i][j] = __builtin_amdgcn_wmma_f32_16x16x32_bf16(
                    false, af, false, bf[j], (short)0, acc[i][j],
                    false, false);
        }
    };

    issueAsync(0, 0);
    asm volatile("s_wait_asynccnt 0x0" ::: "memory");
    __syncthreads();

    for (int kt = 0; kt < NK; ++kt) {
        if (kt + 1 < NK) issueAsync(kt + 1, (kt + 1) & 1);
        compute(kt & 1);
        asm volatile("s_wait_asynccnt 0x0" ::: "memory");
        __syncthreads();
    }

    // ---- epilogue -------------------------------------------------------
    float lsum = 0.0f;
#pragma unroll
    for (int i = 0; i < 4; ++i) {
#pragma unroll
        for (int j = 0; j < 4; ++j) {
            int row0 = blockRow + wm * 64 + i * 16;
            int col  = blockCol + wn * 64 + j * 16 + l16;
#pragma unroll
            for (int vi = 0; vi < 8; ++vi) {
                int row = row0 + vi + 8 * kg;        // C/D layout: M = vi + 8*kg
                size_t idx = (size_t)row * ldc + col;
                float val = acc[i][j][vi];
                if constexpr (EPI == EPI_BF16) {
                    ((unsigned short*)Cout)[idx] = f32_to_bf16(val);
                } else if constexpr (EPI == EPI_F32_BIAS) {
                    ((float*)Cout)[idx] = val + aux[col];
                } else if constexpr (EPI == EPI_GATE) {
                    float gp = aux[idx];
                    ((float*)Cout)[idx] = val / (1.0f + __expf(-gp));
                } else {  // EPI_RECON
                    float d = val - aux[idx];
                    lsum += d * d;
                }
            }
        }
    }
    if constexpr (EPI == EPI_RECON) {
        atomicAdd(lossAccum, lsum);
    }
}

// ---------------------------------------------------------------------------
// prefix mean over P=8 slots: pmean[n,c] = (1/P) sum_p prefix[n,p,c]
// ---------------------------------------------------------------------------
__global__ __launch_bounds__(256)
void prefix_mean_kernel(const unsigned short* __restrict__ prefix,
                        unsigned short* __restrict__ pmean,
                        int N, int C, int P) {
    size_t idx = (size_t)blockIdx.x * blockDim.x + threadIdx.x;
    if (idx >= (size_t)N * C) return;
    size_t n = idx / C;
    int    c = (int)(idx - n * C);
    float s = 0.0f;
    for (int p = 0; p < 8; ++p)
        s += bf16_to_f32(prefix[(n * P + p) * (size_t)C + c]);
    pmean[idx] = f32_to_bf16(s * 0.125f);
}

// ---------------------------------------------------------------------------
// Per-(token, head) attention over P=8 prefix keys, D=128.
// One wave per (n,h); wave32 xor-shuffle reductions for the q.k dots.
// ---------------------------------------------------------------------------
__global__ __launch_bounds__(256)
void attn_kernel(const unsigned short* __restrict__ q,
                 const unsigned short* __restrict__ k,
                 const unsigned short* __restrict__ v,
                 unsigned short* __restrict__ out) {
    const int D = 128, P = 8, C = 2048;
    const int wave = threadIdx.x >> 5;
    const int lane = threadIdx.x & 31;
    const int pairIdx = blockIdx.x * 8 + wave;
    const int n = pairIdx >> 4;
    const int h = pairIdx & 15;

    const size_t qoff = (size_t)n * C + h * D + lane * 4;
    float qv[4];
#pragma unroll
    for (int j = 0; j < 4; ++j) qv[j] = bf16_to_f32(q[qoff + j]);

    float s[8];
#pragma unroll
    for (int p = 0; p < P; ++p) {
        size_t ko = ((size_t)(n * P + p)) * C + h * D + lane * 4;
        float part = 0.0f;
#pragma unroll
        for (int j = 0; j < 4; ++j) part += qv[j] * bf16_to_f32(k[ko + j]);
#pragma unroll
        for (int off = 16; off > 0; off >>= 1)
            part += __shfl_xor(part, off, 32);
        s[p] = part * 0.08838834764831845f;   // 1/sqrt(128)
    }

    float m = s[0];
#pragma unroll
    for (int p = 1; p < P; ++p) m = fmaxf(m, s[p]);
    float e[8], sum = 0.0f;
#pragma unroll
    for (int p = 0; p < P; ++p) { e[p] = __expf(s[p] - m); sum += e[p]; }
    float inv = 1.0f / sum;

    float o[4] = {0.f, 0.f, 0.f, 0.f};
#pragma unroll
    for (int p = 0; p < P; ++p) {
        float a = e[p] * inv;
        size_t vo = ((size_t)(n * P + p)) * C + h * D + lane * 4;
#pragma unroll
        for (int j = 0; j < 4; ++j) o[j] += a * bf16_to_f32(v[vo + j]);
    }
#pragma unroll
    for (int j = 0; j < 4; ++j) out[qoff + j] = f32_to_bf16(o[j]);
}

__global__ void zero_loss_kernel(float* loss) { *loss = 0.0f; }

__global__ void finalize_loss_kernel(const float* loss, float* d_out,
                                     size_t NC) {
    d_out[NC] = (*loss) / (float)NC;
}

// ---------------------------------------------------------------------------
// Host-side orchestration. Workspace plan (~715 MB).
// ---------------------------------------------------------------------------
extern "C" void kernel_launch(void* const* d_in, const int* in_sizes, int n_in,
                              void* d_out, int out_size, void* d_ws, size_t ws_size,
                              hipStream_t stream) {
    (void)in_sizes; (void)n_in; (void)out_size; (void)ws_size;
    const int N = 4096, C = 2048, P = 8, H = 16;
    const int PC = P * C;                  // 16384
    const size_t NC = (size_t)N * C;

    const float* x    = (const float*)d_in[0];
    const float* st   = (const float*)d_in[1];
    const float* Wq   = (const float*)d_in[2];
    const float* Wk   = (const float*)d_in[3];
    const float* Wv   = (const float*)d_in[4];
    const float* Wo   = (const float*)d_in[5];
    const float* Wg   = (const float*)d_in[6];
    const float* bg   = (const float*)d_in[7];
    const float* Wb   = (const float*)d_in[8];
    const float* Wr   = (const float*)d_in[9];
    float* out = (float*)d_out;

    char* ws = (char*)d_ws;
    size_t off = 0;
    auto take = [&](size_t bytes) -> char* {
        char* p = ws + off;
        off += (bytes + 255) & ~(size_t)255;
        return p;
    };
    unsigned short* xs_bf   = (unsigned short*)take((size_t)N * 2 * C * 2);
    unsigned short* x_bf    = (unsigned short*)take(NC * 2);
    unsigned short* Wb_t    = (unsigned short*)take((size_t)2 * C * PC * 2);
    unsigned short* Wq_t    = (unsigned short*)take((size_t)C * C * 2);
    unsigned short* Wk_t    = (unsigned short*)take((size_t)C * C * 2);
    unsigned short* Wv_t    = (unsigned short*)take((size_t)C * C * 2);
    unsigned short* Wo_t    = (unsigned short*)take((size_t)C * C * 2);
    unsigned short* Wg_t    = (unsigned short*)take((size_t)C * C * 2);
    unsigned short* Wr_t    = (unsigned short*)take((size_t)C * C * 2);
    unsigned short* prefix  = (unsigned short*)take((size_t)N * PC * 2);
    unsigned short* kbuf    = (unsigned short*)take((size_t)N * PC * 2);
    unsigned short* vbuf    = (unsigned short*)take((size_t)N * PC * 2);
    unsigned short* qbuf    = (unsigned short*)take(NC * 2);
    unsigned short* pmean   = (unsigned short*)take(NC * 2);
    unsigned short* attnbuf = (unsigned short*)take(NC * 2);
    float*          gpre    = (float*)take(NC * 4);
    float*          loss    = (float*)take(256);

    // 0) conversions + weight transposes
    {
        size_t total = (size_t)N * 2 * C;
        convert_concat_kernel<<<(unsigned)((total + 255) / 256), 256, 0, stream>>>(
            x, st, xs_bf, x_bf, N, C);
    }
    transpose_to_bf16_kernel<<<dim3(PC / 32, (2 * C) / 32), 256, 0, stream>>>(Wb, Wb_t, 2 * C, PC);
    transpose_to_bf16_kernel<<<dim3(C / 32, C / 32), 256, 0, stream>>>(Wq, Wq_t, C, C);
    transpose_to_bf16_kernel<<<dim3(C / 32, C / 32), 256, 0, stream>>>(Wk, Wk_t, C, C);
    transpose_to_bf16_kernel<<<dim3(C / 32, C / 32), 256, 0, stream>>>(Wv, Wv_t, C, C);
    transpose_to_bf16_kernel<<<dim3(C / 32, C / 32), 256, 0, stream>>>(Wo, Wo_t, C, C);
    transpose_to_bf16_kernel<<<dim3(C / 32, C / 32), 256, 0, stream>>>(Wg, Wg_t, C, C);
    transpose_to_bf16_kernel<<<dim3(C / 32, C / 32), 256, 0, stream>>>(Wr, Wr_t, C, C);

    // 1) prefix = xs @ Wb      [4096 x 16384], K = 4096
    gemm_bf16_wmma<EPI_BF16><<<dim3(PC / 256, N / 128), 256, 0, stream>>>(
        xs_bf, 2 * C, Wb_t, 2 * C, prefix, PC, nullptr, nullptr, 2 * C);

    // 2) pmean = mean_p prefix
    prefix_mean_kernel<<<(unsigned)(NC / 256), 256, 0, stream>>>(prefix, pmean, N, C, P);

    // 3) k = prefix @ Wk, v = prefix @ Wv   [32768 x 2048]
    gemm_bf16_wmma<EPI_BF16><<<dim3(C / 256, (N * P) / 128), 256, 0, stream>>>(
        prefix, C, Wk_t, C, kbuf, C, nullptr, nullptr, C);
    gemm_bf16_wmma<EPI_BF16><<<dim3(C / 256, (N * P) / 128), 256, 0, stream>>>(
        prefix, C, Wv_t, C, vbuf, C, nullptr, nullptr, C);

    // 4) q = x @ Wq
    gemm_bf16_wmma<EPI_BF16><<<dim3(C / 256, N / 128), 256, 0, stream>>>(
        x_bf, C, Wq_t, C, qbuf, C, nullptr, nullptr, C);

    // 5) gpre = x @ Wg + bg
    gemm_bf16_wmma<EPI_F32_BIAS><<<dim3(C / 256, N / 128), 256, 0, stream>>>(
        x_bf, C, Wg_t, C, gpre, C, bg, nullptr, C);

    // 6) recon loss
    zero_loss_kernel<<<1, 1, 0, stream>>>(loss);
    gemm_bf16_wmma<EPI_RECON><<<dim3(C / 256, N / 128), 256, 0, stream>>>(
        pmean, C, Wr_t, C, nullptr, C, x, loss, C);

    // 7) attention
    attn_kernel<<<(N * H) / 8, 256, 0, stream>>>(qbuf, kbuf, vbuf, attnbuf);

    // 8) out = (attn @ Wo) * sigmoid(gpre)
    gemm_bf16_wmma<EPI_GATE><<<dim3(C / 256, N / 128), 256, 0, stream>>>(
        attnbuf, C, Wo_t, C, out, C, gpre, nullptr, C);

    // 9) d_out[N*C] = recon_loss
    finalize_loss_kernel<<<1, 1, 0, stream>>>(loss, out, NC);
}
